// TransformerBlock_89696097010056
// MI455X (gfx1250) — matrix operand order repaired and checked by reading, not compiled, not run on hardware
//
#include <hip/hip_runtime.h>
#include <hip/hip_bf16.h>

// ---------------------------------------------------------------------------
// Transformer block for MI455X (gfx1250, wave32, WMMA f32_16x16x32_f16).
// convert weights->f16, LN1, QKV GEMM, flash-attention, out-proj(+bias+resid),
// LN2, MLP1(+bias+GELU), MLP2(+bias+resid) -> d_out.
// ---------------------------------------------------------------------------

typedef __attribute__((ext_vector_type(16))) _Float16       v16h;
typedef __attribute__((ext_vector_type(8)))  _Float16       v8h;
typedef __attribute__((ext_vector_type(8)))  float          v8f;
typedef __attribute__((ext_vector_type(8)))  unsigned short u16x8;
typedef __attribute__((ext_vector_type(4)))  unsigned int   u32x4;
typedef __attribute__((ext_vector_type(4)))  int            i32x4;
typedef __attribute__((ext_vector_type(2)))  unsigned int   u32x2;

#define AS3 __attribute__((address_space(3)))
#define AS1 __attribute__((address_space(1)))

#ifdef __has_builtin
#if __has_builtin(__builtin_amdgcn_global_load_async_to_lds_b128)
#define HAVE_ASYNC_LDS 1
#endif
#if __has_builtin(__builtin_amdgcn_update_dpp)
#define HAVE_DPP 1
#endif
#endif

struct FragU { u32x4 lo, hi; };

// Build a 16x(f16) WMMA fragment from two 16-byte chunks (must be 16B aligned).
static __device__ inline v16h ld_frag(const _Float16* p0, const _Float16* p1) {
  FragU f;
  f.lo = *(const u32x4*)p0;
  f.hi = *(const u32x4*)p1;
  return __builtin_bit_cast(v16h, f);
}

// ---- 16-lane (DPP row) reductions: pure VALU, no ds_bpermute ---------------
#ifdef HAVE_DPP
template <int CTRL>
static __device__ inline float dpp_rot(float x) {
  int i = __builtin_bit_cast(int, x);
  int r = __builtin_amdgcn_update_dpp(i, i, CTRL, 0xf, 0xf, false);
  return __builtin_bit_cast(float, r);
}
static __device__ inline float row_max16(float x) {
  x = fmaxf(x, dpp_rot<0x121>(x));   // row_ror:1
  x = fmaxf(x, dpp_rot<0x122>(x));   // row_ror:2
  x = fmaxf(x, dpp_rot<0x124>(x));   // row_ror:4
  x = fmaxf(x, dpp_rot<0x128>(x));   // row_ror:8
  return x;
}
static __device__ inline float row_sum16(float x) {
  x += dpp_rot<0x121>(x);
  x += dpp_rot<0x122>(x);
  x += dpp_rot<0x124>(x);
  x += dpp_rot<0x128>(x);
  return x;
}
#else
static __device__ inline float row_max16(float x) {
  for (int off = 1; off < 16; off <<= 1) x = fmaxf(x, __shfl_xor(x, off, 32));
  return x;
}
static __device__ inline float row_sum16(float x) {
  for (int off = 1; off < 16; off <<= 1) x += __shfl_xor(x, off, 32);
  return x;
}
#endif

// ---------------------------------------------------------------------------
// fp32 -> f16 elementwise convert (weights)
// ---------------------------------------------------------------------------
__global__ void f32_to_f16(const float* __restrict__ s, _Float16* __restrict__ d, int n) {
  int i = blockIdx.x * blockDim.x + threadIdx.x;
  int stride = gridDim.x * blockDim.x;
  for (; i < n; i += stride) d[i] = (_Float16)s[i];
}

// ---------------------------------------------------------------------------
// LayerNorm over C=1024, one row per block, f16 output
// ---------------------------------------------------------------------------
__global__ __launch_bounds__(256)
void ln_kernel(const float* __restrict__ x, const float* __restrict__ g,
               const float* __restrict__ bt, _Float16* __restrict__ y) {
  __shared__ float r1[256];
  __shared__ float r2[256];
  const int row = blockIdx.x, t = threadIdx.x;
  const float* xb = x + (size_t)row * 1024;
  float s = 0.f, s2 = 0.f;
#pragma unroll
  for (int i = t; i < 1024; i += 256) { float v = xb[i]; s += v; s2 += v * v; }
  r1[t] = s; r2[t] = s2;
  __syncthreads();
  for (int off = 128; off > 0; off >>= 1) {
    if (t < off) { r1[t] += r1[t + off]; r2[t] += r2[t + off]; }
    __syncthreads();
  }
  const float mu   = r1[0] * (1.f / 1024.f);
  const float var  = r2[0] * (1.f / 1024.f) - mu * mu;
  const float rstd = rsqrtf(var + 1e-5f);
#pragma unroll
  for (int i = t; i < 1024; i += 256)
    y[(size_t)row * 1024 + i] = (_Float16)((xb[i] - mu) * rstd * g[i] + bt[i]);
}

// ---------------------------------------------------------------------------
// f16 GEMM: out[M,N] = A[M,K] @ B[K,N] (+bias)(+GELU)(+fp32 residual)
// 256 thr = 8 waves; block tile 128x128; wave tile 32x64 -> 8 WMMA/K-step.
// Register-staged pipeline: next global tile loads overlap current compute.
// LDS: A row-major [128][48], B transposed [128n][48k] (96B rows, 16B aligned).
// ---------------------------------------------------------------------------
#define BM 128
#define BN 128
#define BK 32
#define LDA 48
#define LDB 48

template<bool BIAS, bool GELU_ACT, bool RESID, bool OUTF16>
__global__ __launch_bounds__(256)
void gemm_f16(const _Float16* __restrict__ A, const _Float16* __restrict__ Bw,
              const float* __restrict__ bias, const float* __restrict__ resid,
              void* __restrict__ outp, int M, int N, int K) {
  __shared__ __attribute__((aligned(16))) _Float16 lA[BM * LDA];
  __shared__ __attribute__((aligned(16))) _Float16 lB[BN * LDB];
  const int t    = threadIdx.x;
  const int lane = t & 31, wave = t >> 5;
  const int half = lane >> 4, l15 = lane & 15;
  const int wm = wave >> 1, wn = wave & 1;      // 4x2 waves -> 128x128 tile
  const int m0 = blockIdx.y * BM, n0 = blockIdx.x * BN;

  // staging coordinates
  const int ar = t >> 2;          // 0..63   A rows (and +64)
  const int ac = (t & 3) * 8;     // 0,8,16,24
  const int bk = (t >> 4) * 2;    // 0..30 even k
  const int bn = (t & 15) * 8;    // 0..120

  v8f acc[2][4];
#pragma unroll
  for (int i = 0; i < 2; ++i)
#pragma unroll
    for (int j = 0; j < 4; ++j)
#pragma unroll
      for (int e = 0; e < 8; ++e) acc[i][j][e] = 0.f;

  u32x4 sA0, sA1, sB0, sB1;
  auto load_tile = [&](int k0) {
    sA0 = *(const u32x4*)&A[(size_t)(m0 + ar) * K + k0 + ac];
    sA1 = *(const u32x4*)&A[(size_t)(m0 + ar + 64) * K + k0 + ac];
    sB0 = *(const u32x4*)&Bw[(size_t)(k0 + bk) * N + n0 + bn];
    sB1 = *(const u32x4*)&Bw[(size_t)(k0 + bk + 1) * N + n0 + bn];
  };
  load_tile(0);

  for (int k0 = 0; k0 < K; k0 += BK) {
    // commit staged tile to LDS
    *(u32x4*)&lA[ar * LDA + ac]        = sA0;
    *(u32x4*)&lA[(ar + 64) * LDA + ac] = sA1;
    {
      u16x8 h0 = __builtin_bit_cast(u16x8, sB0);
      u16x8 h1 = __builtin_bit_cast(u16x8, sB1);
#pragma unroll
      for (int i = 0; i < 8; ++i) {
        unsigned int v = (unsigned int)h0[i] | ((unsigned int)h1[i] << 16);
        *(unsigned int*)&lB[(bn + i) * LDB + bk] = v;   // transposed [n][k]
      }
    }
    __syncthreads();

    // prefetch next K tile into registers (overlaps WMMA below)
    int kn = k0 + BK;
    if (kn >= K) kn = 0;
    load_tile(kn);

    v16h aF[2], bF[4];
#pragma unroll
    for (int mt = 0; mt < 2; ++mt) {
      const _Float16* rb = &lA[(wm * 32 + mt * 16 + l15) * LDA];
      aF[mt] = ld_frag(rb + half * 8, rb + 16 + half * 8);   // A layout V0-3 / V4-7
    }
#pragma unroll
    for (int nt = 0; nt < 4; ++nt) {
      const _Float16* rb = &lB[(wn * 64 + nt * 16 + l15) * LDB];
      bF[nt] = ld_frag(rb + half * 8, rb + 16 + half * 8);   // B layout V0-3 / V4-7 (same k order as A)
    }
#pragma unroll
    for (int mt = 0; mt < 2; ++mt)
#pragma unroll
      for (int nt = 0; nt < 4; ++nt)
        acc[mt][nt] = __builtin_amdgcn_wmma_f32_16x16x32_f16(
            false, aF[mt], false, bF[nt], (short)0, acc[mt][nt], false, false);
    __syncthreads();
  }

  // Epilogue per C/D layout: lane -> N=l15, VGPR e -> M = e + 8*half
#pragma unroll
  for (int mt = 0; mt < 2; ++mt)
#pragma unroll
    for (int nt = 0; nt < 4; ++nt)
#pragma unroll
      for (int e = 0; e < 8; ++e) {
        const int m = m0 + wm * 32 + mt * 16 + e + 8 * half;
        const int n = n0 + wn * 64 + nt * 16 + l15;
        float v = acc[mt][nt][e];
        if (BIAS)     v += bias[n];
        if (GELU_ACT) v = 0.5f * v * (1.0f + erff(v * 0.70710678118654752f));
        if (RESID)    v += resid[(size_t)m * N + n];
        if (OUTF16) ((_Float16*)outp)[(size_t)m * N + n] = (_Float16)v;
        else        ((float*)outp)[(size_t)m * N + n]    = v;
      }
}

// ---------------------------------------------------------------------------
// Flash attention: grid (T/64, B*NH), 128 thr = 4 waves, 16 q-rows/wave,
// key blocks of 32. qkv row = [q(1024) | k(1024) | v(1024)], head h at h*64.
// ---------------------------------------------------------------------------
#define NHEADS 16
#define TSEQ   2048
#define DH     64

__global__ __launch_bounds__(128)
void attn_kernel(const _Float16* __restrict__ qkv, _Float16* __restrict__ o) {
  __shared__ __attribute__((aligned(16))) _Float16 lK[32 * 72];      // [key][d], 144B rows
  __shared__ __attribute__((aligned(16))) _Float16 lVt[64 * 48];     // [d][key], 96B rows
  __shared__ __attribute__((aligned(16))) _Float16 lP[4 * 16 * 48];  // per-wave P tile
  const int t    = threadIdx.x;
  const int lane = t & 31, wave = t >> 5;
  const int half = lane >> 4, l15 = lane & 15;
  const int bh = blockIdx.y;
  const int b = bh >> 4, h = bh & 15;
  const int qb = blockIdx.x * 64;
  const int q0 = qb + wave * 16;
  const size_t rs = 3072;
  const _Float16* base = qkv + (size_t)(b * TSEQ) * rs;

  // Q fragments (16 rows x 64 d) straight from global per the A layout.
  v16h aQ[2];
  {
    const _Float16* rq = base + (size_t)(q0 + l15) * rs + h * DH;
#pragma unroll
    for (int c = 0; c < 2; ++c)
      aQ[c] = ld_frag(rq + c * 32 + half * 8, rq + c * 32 + 16 + half * 8);
  }

  float mrun[8], lrun[8];
  v8f accO[4];
#pragma unroll
  for (int e = 0; e < 8; ++e) { mrun[e] = -1e30f; lrun[e] = 0.f; }
#pragma unroll
  for (int d = 0; d < 4; ++d)
#pragma unroll
    for (int e = 0; e < 8; ++e) accO[d][e] = 0.f;

  // staging coords
  const int skey = t & 31;           // K tile: one key row per thread
  const int sdb  = (t >> 5) * 16;    // 16 d per thread
  const int vj   = (t & 15) * 2;     // V tile: even key pair
  const int vdb  = (t >> 4) * 8;     // 8 d per thread

  const int nblocks = (qb + 64) / 32;
  for (int jb = 0; jb < nblocks; ++jb) {
    const int kb = jb * 32;
    // ---- stage K [32][64] (async direct-to-LDS when available) ----
    {
      const _Float16* kg = base + (size_t)(kb + skey) * rs + 1024 + h * DH + sdb;
#ifdef HAVE_ASYNC_LDS
      __builtin_amdgcn_global_load_async_to_lds_b128(
          (i32x4*)kg, (AS3 i32x4*)&lK[skey * 72 + sdb], 0, 0);
      __builtin_amdgcn_global_load_async_to_lds_b128(
          (i32x4*)(kg + 8), (AS3 i32x4*)&lK[skey * 72 + sdb + 8], 0, 0);
#else
      *(u32x4*)&lK[skey * 72 + sdb]     = *(const u32x4*)kg;
      *(u32x4*)&lK[skey * 72 + sdb + 8] = *(const u32x4*)(kg + 8);
#endif
    }
    // ---- stage V transposed [64][32], packed b32 stores ----
    {
      const _Float16* vg0 = base + (size_t)(kb + vj) * rs + 2048 + h * DH + vdb;
      const _Float16* vg1 = vg0 + rs;
      u16x8 h0 = __builtin_bit_cast(u16x8, *(const u32x4*)vg0);
      u16x8 h1 = __builtin_bit_cast(u16x8, *(const u32x4*)vg1);
#pragma unroll
      for (int i = 0; i < 8; ++i) {
        unsigned int v = (unsigned int)h0[i] | ((unsigned int)h1[i] << 16);
        *(unsigned int*)&lVt[(vdb + i) * 48 + vj] = v;
      }
    }
#ifdef HAVE_ASYNC_LDS
#if __has_builtin(__builtin_amdgcn_s_wait_asynccnt)
    __builtin_amdgcn_s_wait_asynccnt(0);
#else
    asm volatile("s_wait_asynccnt 0" ::: "memory");
#endif
#endif
    __syncthreads();

    if (kb <= q0 + 15) {   // wave-uniform: EXEC all-ones inside (WMMA-safe)
      // S = Q K^T : two 16x16 key tiles, K-dim 64 -> 2 WMMAs each
      v8f s[2];
#pragma unroll
      for (int nt = 0; nt < 2; ++nt) {
#pragma unroll
        for (int e = 0; e < 8; ++e) s[nt][e] = 0.f;
        const _Float16* rb = &lK[(nt * 16 + l15) * 72];
#pragma unroll
        for (int c = 0; c < 2; ++c) {
          v16h bK = ld_frag(rb + c * 32 + half * 8, rb + c * 32 + 16 + half * 8);
          s[nt] = __builtin_amdgcn_wmma_f32_16x16x32_f16(
              false, aQ[c], false, bK, (short)0, s[nt], false, false);
        }
      }
      // Online softmax (row = e + 8*half, key col = l15 across lanes)
      float p[2][8], corr[8];
#pragma unroll
      for (int e = 0; e < 8; ++e) {
        const int qrow = q0 + e + 8 * half;
        float s0 = s[0][e] * 0.125f;   // 1/sqrt(64)
        float s1 = s[1][e] * 0.125f;
        if (kb + l15 > qrow)      s0 = -1e30f;
        if (kb + 16 + l15 > qrow) s1 = -1e30f;
        const float rmax = row_max16(fmaxf(s0, s1));
        const float mnew = fmaxf(mrun[e], rmax);
        corr[e] = expf(mrun[e] - mnew);
        const float p0 = expf(s0 - mnew), p1 = expf(s1 - mnew);
        const float rsum = row_sum16(p0 + p1);
        lrun[e] = lrun[e] * corr[e] + rsum;
        mrun[e] = mnew;
        p[0][e] = p0; p[1][e] = p1;
      }
#pragma unroll
      for (int d = 0; d < 4; ++d)
#pragma unroll
        for (int e = 0; e < 8; ++e) accO[d][e] *= corr[e];

      // Re-layout P (C layout -> A fragment) through per-wave LDS.
      _Float16* lPw = &lP[wave * 16 * 48];
#pragma unroll
      for (int nt = 0; nt < 2; ++nt)
#pragma unroll
        for (int e = 0; e < 8; ++e)
          lPw[(e + 8 * half) * 48 + nt * 16 + l15] = (_Float16)p[nt][e];
      asm volatile("s_wait_dscnt 0" ::: "memory");  // per-wave LDS store->load
      {
        const _Float16* rp = &lPw[l15 * 48];
        v16h aP = ld_frag(rp + half * 8, rp + 16 + half * 8);
#pragma unroll
        for (int d = 0; d < 4; ++d) {
          const _Float16* rv = &lVt[(d * 16 + l15) * 48];
          v16h bV = ld_frag(rv + half * 8, rv + 16 + half * 8);
          accO[d] = __builtin_amdgcn_wmma_f32_16x16x32_f16(
              false, aP, false, bV, (short)0, accO[d], false, false);
        }
      }
    }
    __syncthreads();
  }

  // Normalize and write O (f16) in interleaved-head layout [B*T][1024].
#pragma unroll
  for (int d = 0; d < 4; ++d)
#pragma unroll
    for (int e = 0; e < 8; ++e) {
      const int qrow = q0 + e + 8 * half;
      const int col  = h * DH + d * 16 + l15;
      o[(size_t)(b * TSEQ + qrow) * 1024 + col] = (_Float16)(accO[d][e] / lrun[e]);
    }
}

// ---------------------------------------------------------------------------
// Host-side launcher
// ---------------------------------------------------------------------------
extern "C" void kernel_launch(void* const* d_in, const int* in_sizes, int n_in,
                              void* d_out, int out_size, void* d_ws, size_t ws_size,
                              hipStream_t stream) {
  (void)in_sizes; (void)n_in; (void)out_size; (void)ws_size;
  const float* x     = (const float*)d_in[0];
  const float* ln1_g = (const float*)d_in[1];
  const float* ln1_b = (const float*)d_in[2];
  const float* w_qkv = (const float*)d_in[3];
  const float* w_out = (const float*)d_in[4];
  const float* b_out = (const float*)d_in[5];
  const float* ln2_g = (const float*)d_in[6];
  const float* ln2_b = (const float*)d_in[7];
  const float* w1    = (const float*)d_in[8];
  const float* b1    = (const float*)d_in[9];
  const float* w2    = (const float*)d_in[10];
  const float* b2    = (const float*)d_in[11];

  const int M  = 4 * 2048;   // B*T rows
  const int C  = 1024, C3 = 3 * 1024, F = 4 * 1024;

  // Workspace carve-out (all 256B aligned). ff aliases qkv+o (exact fit),
  // h2 aliases h. Total ~142 MB.
  char* ws = (char*)d_ws;
  size_t off = 0;
  auto carve = [&](size_t bytes) -> char* {
    char* p = ws + off;
    off += (bytes + 255) & ~(size_t)255;
    return p;
  };
  _Float16* wqkv_h = (_Float16*)carve((size_t)C * C3 * 2);
  _Float16* wout_h = (_Float16*)carve((size_t)C * C  * 2);
  _Float16* w1_h   = (_Float16*)carve((size_t)C * F  * 2);
  _Float16* w2_h   = (_Float16*)carve((size_t)F * C  * 2);
  _Float16* h_h    = (_Float16*)carve((size_t)M * C  * 2);
  _Float16* qkv_h  = (_Float16*)carve((size_t)M * C3 * 2);
  _Float16* o_h    = (_Float16*)carve((size_t)M * C  * 2);
  float*    x2_f   = (float*)   carve((size_t)M * C  * 4);
  _Float16* ff_h   = qkv_h;   // 8192x4096 reuses qkv(48MB)+o(16MB) exactly
  _Float16* h2_h   = h_h;

  // 1) Weights -> f16
  f32_to_f16<<<dim3(512), dim3(256), 0, stream>>>(w_qkv, wqkv_h, C * C3);
  f32_to_f16<<<dim3(512), dim3(256), 0, stream>>>(w_out, wout_h, C * C);
  f32_to_f16<<<dim3(512), dim3(256), 0, stream>>>(w1,    w1_h,   C * F);
  f32_to_f16<<<dim3(512), dim3(256), 0, stream>>>(w2,    w2_h,   F * C);

  // 2) LN1
  ln_kernel<<<dim3(M), dim3(256), 0, stream>>>(x, ln1_g, ln1_b, h_h);

  // 3) QKV = h @ w_qkv  (f16 out, no bias)
  gemm_f16<false, false, false, true>
      <<<dim3(C3 / BN, M / BM), dim3(256), 0, stream>>>(h_h, wqkv_h, nullptr, nullptr, qkv_h, M, C3, C);

  // 4) Flash attention -> o (f16)
  attn_kernel<<<dim3(TSEQ / 64, 4 * NHEADS), dim3(128), 0, stream>>>(qkv_h, o_h);

  // 5) x2 = x + o @ w_out + b_out  (fp32 out)
  gemm_f16<true, false, true, false>
      <<<dim3(C / BN, M / BM), dim3(256), 0, stream>>>(o_h, wout_h, b_out, x, x2_f, M, C, C);

  // 6) LN2
  ln_kernel<<<dim3(M), dim3(256), 0, stream>>>(x2_f, ln2_g, ln2_b, h2_h);

  // 7) ff = GELU(h2 @ w1 + b1)  (f16 out)
  gemm_f16<true, true, false, true>
      <<<dim3(F / BN, M / BM), dim3(256), 0, stream>>>(h2_h, w1_h, b1, nullptr, ff_h, M, F, C);

  // 8) out = x2 + ff @ w2 + b2  (fp32 out)
  gemm_f16<true, false, true, false>
      <<<dim3(C / BN, M / BM), dim3(256), 0, stream>>>(ff_h, w2_h, b2, x2_f, (float*)d_out, M, C, F);
}
